// Encoder_61572651155777
// MI455X (gfx1250) — matrix operand order repaired and checked by reading, not compiled, not run on hardware
//
#include <hip/hip_runtime.h>
#include <hip/hip_bf16.h>

// Problem dims (fixed by the reference)
#define BB 512
#define TT 512
#define EE 50
#define HH 64
#define GG 256   // 4*H

typedef __attribute__((ext_vector_type(16))) __bf16        v16bf;
typedef __attribute__((ext_vector_type(8)))  float         v8f;
typedef __attribute__((ext_vector_type(8)))  unsigned int  v8u;

// f32 -> bf16 (round to nearest even), returned in low 16 bits
__device__ __forceinline__ unsigned f2bf(float x) {
    unsigned u = __builtin_bit_cast(unsigned, x);
    return (u + 0x7FFFu + ((u >> 16) & 1u)) >> 16;
}

// Branchless tanh: hardware v_tanh_f32 if the builtin exists, else exp2+rcp
// (tanh(x) = 1 - 2/(exp(2x)+1); saturates correctly via inf/0 at the extremes).
__device__ __forceinline__ float fast_tanh(float x) {
#if __has_builtin(__builtin_amdgcn_tanhf)
    return __builtin_amdgcn_tanhf(x);
#else
    const float e = __builtin_amdgcn_exp2f(x * 2.885390081777927f); // 2*log2(e)
    return 1.0f - 2.0f * __builtin_amdgcn_rcpf(e + 1.0f);
#endif
}

// sigmoid(x) = 0.5*tanh(x/2) + 0.5  -> single trans op on CDNA5
__device__ __forceinline__ float fast_sigmoid(float x) {
    return __builtin_fmaf(0.5f, fast_tanh(0.5f * x), 0.5f);
}

// Load one 32x16 bf16 B-fragment (K x N) from a row-major f32 matrix M[k*GG + n],
// zero-padding rows k >= kmax. Contiguous-k layout (matches A): lane L -> N = L&15, K-half = (L<16?0:16),
// VGPR v holds packed bf16 pair (K = half+2v, half+2v+1).
__device__ __forceinline__ v8u load_bfrag(const float* __restrict__ M,
                                          int kbase, int kmax, int n0, int lane) {
    const int n    = n0 + (lane & 15);
    const int koff = (lane < 16) ? 0 : 16;
    v8u r;
#pragma unroll
    for (int v = 0; v < 8; ++v) {
        const int k = kbase + koff + 2 * v;
        unsigned lo = (k     < kmax) ? f2bf(M[(k    ) * GG + n]) : 0u;
        unsigned hi = (k + 1 < kmax) ? f2bf(M[(k + 1) * GG + n]) : 0u;
        r[v] = lo | (hi << 16);
    }
    return r;
}

// Load one 16x32 bf16 A-fragment from a row-major LDS tile of ushort [16][64],
// selecting K-chunk `kchunk` (0 -> K 0..31, 1 -> K 32..63).
// Contiguous-k layout (matches B): lane L -> M = L&15; kb = (L<16?0:16);
// VGPR v holds K pair k0 = chunk*32 + kb + 2*v.
__device__ __forceinline__ v8u load_afrag(const unsigned short* lds, int kchunk, int lane) {
    const int m  = lane & 15;
    const int kb = (lane < 16) ? 0 : 16;
    v8u r;
#pragma unroll
    for (int v = 0; v < 8; ++v) {
        const int k0 = kchunk * 32 + kb + 2 * v; // even -> dword aligned
        r[v] = *(const unsigned*)(lds + m * 64 + k0);
    }
    return r;
}

__device__ __forceinline__ v8f wmma_bf16(v8u a, v8u b, v8f c) {
    return __builtin_amdgcn_wmma_f32_16x16x32_bf16(
        /*neg_a=*/false, __builtin_bit_cast(v16bf, a),
        /*neg_b=*/false, __builtin_bit_cast(v16bf, b),
        /*c_mod=*/(short)0, c, /*reuse_a=*/false, /*reuse_b=*/false);
}

__global__ __launch_bounds__(256, 1)
void lstm_encoder_fused(const int*   __restrict__ tokens,
                        const float* __restrict__ h0,
                        const float* __restrict__ c0,
                        const float* __restrict__ emb,
                        const float* __restrict__ W,
                        const float* __restrict__ U,
                        const float* __restrict__ bias,
                        float*       __restrict__ out) {
    __shared__ unsigned short lds_x[16 * 64];  // bf16 x_t tile (E=50 padded to 64)
    __shared__ unsigned short lds_h[16 * 64];  // bf16 h_{t-1} tile
    __shared__ float          lds_z[16 * 256]; // fp32 pre-activation z

    const int tid  = threadIdx.x;
    const int lane = tid & 31;
    const int wave = tid >> 5;
    const int b0   = blockIdx.x * 16;          // batch-tile base

    // ---- gate-phase ownership: thread -> (row, 4 consecutive hidden units) ----
    const int grow = tid >> 4;                 // 0..15
    const int gj0  = (tid & 15) * 4;           // 0..60
    float hreg[4], creg[4], preg[4];
#pragma unroll
    for (int q = 0; q < 4; ++q) {
        hreg[q] = h0[(b0 + grow) * HH + gj0 + q];
        creg[q] = c0[(b0 + grow) * HH + gj0 + q];
        preg[q] = 0.0f;
        lds_h[grow * 64 + gj0 + q] = (unsigned short)f2bf(hreg[q]);
    }

    // ---- hoist loop-invariant B fragments (W with K<50 zero-padded, U dense) ----
    const int n0a = (wave * 2)     * 16;
    const int n0b = (wave * 2 + 1) * 16;
    v8u wfa0 = load_bfrag(W,  0, EE, n0a, lane), wfa1 = load_bfrag(W, 32, EE, n0a, lane);
    v8u wfb0 = load_bfrag(W,  0, EE, n0b, lane), wfb1 = load_bfrag(W, 32, EE, n0b, lane);
    v8u ufa0 = load_bfrag(U,  0, HH, n0a, lane), ufa1 = load_bfrag(U, 32, HH, n0a, lane);
    v8u ufb0 = load_bfrag(U,  0, HH, n0b, lane), ufb1 = load_bfrag(U, 32, HH, n0b, lane);
    const float bna = bias[n0a + (lane & 15)];
    const float bnb = bias[n0b + (lane & 15)];

    __syncthreads();

    for (int t = 0; t < TT; ++t) {
        // ---- stage x_t = emb[tokens[b0+row, t]] into LDS as bf16, padded to K=64.
        //      Clamped-index load + select: no EXEC divergence. ----
        {
            const int row = tid >> 4;
            const int l   = tid & 15;
            const int tok = tokens[(b0 + row) * TT + t];
            const float* er = emb + (long)tok * EE;
#pragma unroll
            for (int k = 0; k < 4; ++k) {
                const int  kk   = l + k * 16;
                const bool ok   = (kk < EE);
                float v = er[ok ? kk : 0];
                v = ok ? v : 0.0f;
                lds_x[row * 64 + kk] = (unsigned short)f2bf(v);
            }
        }
        __syncthreads();   // lds_x ready; lds_h from previous gate phase ready

        // ---- z = [x | h] @ [W; U] : 4 K-chunks of 16x16x32 bf16 WMMA per N-tile.
        //      Bias folded into accumulator init. ----
        {
            v8u ax0 = load_afrag(lds_x, 0, lane);
            v8u ax1 = load_afrag(lds_x, 1, lane);
            v8u ah0 = load_afrag(lds_h, 0, lane);
            v8u ah1 = load_afrag(lds_h, 1, lane);
            v8f acca, accb;
#pragma unroll
            for (int r = 0; r < 8; ++r) { acca[r] = bna; accb[r] = bnb; }
            acca = wmma_bf16(ax0, wfa0, acca);
            acca = wmma_bf16(ax1, wfa1, acca);
            acca = wmma_bf16(ah0, ufa0, acca);
            acca = wmma_bf16(ah1, ufa1, acca);
            accb = wmma_bf16(ax0, wfb0, accb);
            accb = wmma_bf16(ax1, wfb1, accb);
            accb = wmma_bf16(ah0, ufb0, accb);
            accb = wmma_bf16(ah1, ufb1, accb);
            // C/D layout: VGPR r -> M = r (lanes 0-15) or 8+r (lanes 16-31), N = lane&15
            const int mo = (lane < 16) ? 0 : 8;
            const int nn = lane & 15;
#pragma unroll
            for (int r = 0; r < 8; ++r) {
                lds_z[(r + mo) * 256 + n0a + nn] = acca[r];
                lds_z[(r + mo) * 256 + n0b + nn] = accb[r];
            }
        }
        __syncthreads();   // lds_z ready; all waves done reading lds_h

        // ---- gate math (branchless trans ops), state update, masking, output ----
        {
            const int  tok = tokens[(b0 + grow) * TT + t];
            const bool msk = (tok != 0);
            float* orow = out + ((long)(b0 + grow) * TT + t) * HH;
#pragma unroll
            for (int q = 0; q < 4; ++q) {
                const int j  = gj0 + q;
                const float zi = lds_z[grow * 256 +        j];
                const float zf = lds_z[grow * 256 +  64 + j];
                const float zg = lds_z[grow * 256 + 128 + j];
                const float zo = lds_z[grow * 256 + 192 + j];
                const float iv = fast_sigmoid(zi);
                const float fv = fast_sigmoid(zf);
                const float ov = fast_sigmoid(zo);
                const float gv = fast_tanh(zg);
                const float cn = fv * creg[q] + iv * gv;
                const float hn = ov * fast_tanh(cn);
                hreg[q] = msk ? hn : hreg[q];
                creg[q] = msk ? cn : creg[q];
                preg[q] = msk ? hn : preg[q];
                __builtin_nontemporal_store(preg[q], &orow[j]);
                lds_h[grow * 64 + j] = (unsigned short)f2bf(hreg[q]);
            }
        }
        __syncthreads();   // gate phase done before next iteration's WMMA reads lds_h
    }

    // ---- final states: hT at out + B*T*H, cT right after ----
    float* hT = out + (long)BB * TT * HH;
    float* cT = hT + (long)BB * HH;
#pragma unroll
    for (int q = 0; q < 4; ++q) {
        hT[(b0 + grow) * HH + gj0 + q] = hreg[q];
        cT[(b0 + grow) * HH + gj0 + q] = creg[q];
    }
}

extern "C" void kernel_launch(void* const* d_in, const int* in_sizes, int n_in,
                              void* d_out, int out_size, void* d_ws, size_t ws_size,
                              hipStream_t stream) {
    (void)in_sizes; (void)n_in; (void)out_size; (void)d_ws; (void)ws_size;
    const int*   tokens = (const int*)  d_in[0];
    const float* h0     = (const float*)d_in[1];
    const float* c0     = (const float*)d_in[2];
    const float* emb    = (const float*)d_in[3];
    const float* W      = (const float*)d_in[4];
    const float* U      = (const float*)d_in[5];
    const float* b      = (const float*)d_in[6];
    float*       out    = (float*)d_out;

    hipLaunchKernelGGL(lstm_encoder_fused, dim3(BB / 16), dim3(256), 0, stream,
                       tokens, h0, c0, emb, W, U, b, out);
}
